// ScannCore_5446018531523
// MI455X (gfx1250) — matrix operand order repaired and checked by reading, not compiled, not run on hardware
//
#include <hip/hip_runtime.h>
#include <hip/hip_bf16.h>

// Problem constants
#define BB  32
#define INC 2048
#define SS  512

typedef __attribute__((ext_vector_type(16))) __bf16          bf16x16;
typedef __attribute__((ext_vector_type(8)))  float           f32x8;
typedef __attribute__((ext_vector_type(4)))  float           f32x4;
typedef __attribute__((ext_vector_type(4)))  unsigned short  u16x4;
typedef __attribute__((ext_vector_type(8)))  unsigned short  u16x8;
typedef __attribute__((ext_vector_type(16))) unsigned short  u16x16;

// ---- gfx1250 async global->LDS path (ASYNCcnt), with safe fallback --------
#if defined(__gfx1250__) && __has_builtin(__builtin_amdgcn_global_load_async_to_lds_b128)
#define HAVE_ASYNC 1
#else
#define HAVE_ASYNC 0
#endif

#if HAVE_ASYNC
// exact pointee type per clang's declared signature:
//   void (*)(v4i __device__*, v4i __shared__*, imm int, imm int)
typedef int v4i_raw __attribute__((__vector_size__(4 * sizeof(int))));
typedef __attribute__((address_space(1))) v4i_raw gas_v4i;   // global
typedef __attribute__((address_space(3))) v4i_raw las_v4i;   // LDS
#endif

// copy 16 bytes (8 bf16) global -> LDS
__device__ __forceinline__ void cp16_g2l(unsigned short* dst_lds,
                                         const unsigned short* src_glb) {
#if HAVE_ASYNC
    __builtin_amdgcn_global_load_async_to_lds_b128(
        (gas_v4i*)src_glb, (las_v4i*)dst_lds, 0, 0);
#else
    *(u16x8*)dst_lds = *(const u16x8*)src_glb;
#endif
}

template <int N>
__device__ __forceinline__ void async_wait() {
#if HAVE_ASYNC
#if __has_builtin(__builtin_amdgcn_s_wait_asynccnt)
    __builtin_amdgcn_s_wait_asynccnt((short)N);
#else
    asm volatile("s_wait_asynccnt %0" ::"i"(N) : "memory");
#endif
#endif
}

__device__ __forceinline__ unsigned short f2bf(float f) {
    unsigned int u = __builtin_bit_cast(unsigned int, f);
    u += 0x7FFFu + ((u >> 16) & 1u);           // round-to-nearest-even
    return (unsigned short)(u >> 16);
}
__device__ __forceinline__ float bf2f(unsigned short h) {
    unsigned int u = ((unsigned int)h) << 16;
    return __builtin_bit_cast(float, u);
}

// A-fragment (16x32 bf16, ISA 7.12.2): lane half = lane>>4, m = lane&15.
// elems j=0..15 -> K = half*16 + j  (contiguous 16: this code's order, not the ISA table; B matches)
__device__ __forceinline__ bf16x16 load_frag_a(const unsigned short* row, int half) {
    u16x16 c = *(const u16x16*)(row + half * 16);
    return __builtin_bit_cast(bf16x16, c);
}
// B-fragment (32x16 bf16) staged in LDS as [N][K]: lane n = lane&15 picks row,
// elems j=0..15 -> K = (lane>>4)*16 + j  (contiguous 16)
__device__ __forceinline__ bf16x16 load_frag_b(const unsigned short* nrow, int half) {
    u16x16 c = *(const u16x16*)(nrow + half * 16);
    return __builtin_bit_cast(bf16x16, c);
}

// ---------------------------------------------------------------------------
// Kernel 0: f32 -> bf16 bulk convert (x and the three weight matrices)
// ---------------------------------------------------------------------------
__global__ __launch_bounds__(256) void cvt_kernel(
    const float* __restrict__ s, unsigned short* __restrict__ d)
{
    const size_t i = ((size_t)blockIdx.x * 256 + threadIdx.x) * 4;
    f32x4 v = *(const f32x4*)(s + i);
    u16x4 o;
#pragma unroll
    for (int j = 0; j < 4; ++j) o[j] = f2bf(v[j]);
    *(u16x4*)(d + i) = o;
}

// ---------------------------------------------------------------------------
// Kernel 1: Q/K/V = x @ W^T + b (all bf16 operands), written TRANSPOSED:
// Xt[b][s][i] bf16. Block tile M=128 (INC), N=64 (S), K=32, double-buffered
// async staging. 8 waves: wm 0..3, wn 0..1; 2x2 16x16 tiles per weight.
// ---------------------------------------------------------------------------
__global__ __launch_bounds__(256) void qkv_kernel(
    const unsigned short* __restrict__ xb,
    const unsigned short* __restrict__ Wqb,
    const unsigned short* __restrict__ Wkb,
    const unsigned short* __restrict__ Wvb,
    const float* __restrict__ bq, const float* __restrict__ bk,
    const float* __restrict__ bv,
    unsigned short* __restrict__ Qt, unsigned short* __restrict__ Kt,
    unsigned short* __restrict__ Vt)
{
    const int b   = blockIdx.z;
    const int m0  = blockIdx.y * 128;   // INC tile
    const int n0  = blockIdx.x * 64;    // S tile
    const int tid = threadIdx.x;
    const int lane = tid & 31, wave = tid >> 5;
    const int wm = wave & 3, wn = wave >> 2;
    const int half = lane >> 4, nl = lane & 15;

    __shared__ unsigned short As[2][128 * 32];
    __shared__ unsigned short Bs[2][3][64 * 32];

    // staging coordinates
    const int rA = tid >> 1, cA = (tid & 1) * 16;   // 128 x 32, 16 elems/thread
    const int rB = tid >> 2, cB = (tid & 3) * 8;    // 64 x 32, 8 elems/thread
    const unsigned short* srcA =
        xb + (size_t)b * INC * SS + (size_t)(m0 + rA) * SS + cA;
    const unsigned short* srcB[3] = {
        Wqb + (size_t)(n0 + rB) * SS + cB,
        Wkb + (size_t)(n0 + rB) * SS + cB,
        Wvb + (size_t)(n0 + rB) * SS + cB };

    f32x8 acc[3][2][2] = {};

    // prologue: stage k0 = 0 into buffer 0
    cp16_g2l(&As[0][rA * 32 + cA],     srcA);
    cp16_g2l(&As[0][rA * 32 + cA + 8], srcA + 8);
#pragma unroll
    for (int w = 0; w < 3; ++w)
        cp16_g2l(&Bs[0][w][rB * 32 + cB], srcB[w]);

    for (int k0 = 0; k0 < SS; k0 += 32) {
        const int buf = (k0 >> 5) & 1;
        if (k0 + 32 < SS) {
            const int nb = buf ^ 1;
            cp16_g2l(&As[nb][rA * 32 + cA],     srcA + k0 + 32);
            cp16_g2l(&As[nb][rA * 32 + cA + 8], srcA + k0 + 40);
#pragma unroll
            for (int w = 0; w < 3; ++w)
                cp16_g2l(&Bs[nb][w][rB * 32 + cB], srcB[w] + k0 + 32);
            async_wait<5>();   // current buffer's 5 per-wave copies are done
        } else {
            async_wait<0>();
        }
        __syncthreads();

        bf16x16 afrag[2];
#pragma unroll
        for (int mi = 0; mi < 2; ++mi)
            afrag[mi] = load_frag_a(&As[buf][(wm * 32 + mi * 16 + nl) * 32], half);
#pragma unroll
        for (int w = 0; w < 3; ++w) {
#pragma unroll
            for (int ni = 0; ni < 2; ++ni) {
                bf16x16 bfrag =
                    load_frag_b(&Bs[buf][w][(wn * 32 + ni * 16 + nl) * 32], half);
#pragma unroll
                for (int mi = 0; mi < 2; ++mi)
                    acc[w][mi][ni] = __builtin_amdgcn_wmma_f32_16x16x32_bf16(
                        false, afrag[mi], false, bfrag, (short)0,
                        acc[w][mi][ni], false, false);
            }
        }
        __syncthreads();
    }

    // Epilogue: D layout -> lane owns column n (= s), 8 consecutive m (= i).
    const float* biases[3] = { bq, bk, bv };
    unsigned short* outs[3] = { Qt, Kt, Vt };
#pragma unroll
    for (int w = 0; w < 3; ++w) {
#pragma unroll
        for (int ni = 0; ni < 2; ++ni) {
            const int s = n0 + wn * 32 + ni * 16 + nl;
            const float bias = biases[w][s];
#pragma unroll
            for (int mi = 0; mi < 2; ++mi) {
                const int i = m0 + wm * 32 + mi * 16 + half * 8;
                u16x8 pack;
#pragma unroll
                for (int r = 0; r < 8; ++r) pack[r] = f2bf(acc[w][mi][ni][r] + bias);
                *(u16x8*)(outs[w] + ((size_t)b * SS + s) * INC + i) = pack;
            }
        }
    }
}

// ---------------------------------------------------------------------------
// Kernel 2: Et[b][t][s] = E[b][s][t] = sum_i K[b,i,s] * Q[b,i,t]
// A = Kt rows (M=s, K=i), B = Qt rows (N=t, K=i). M=128, N=64, K-step 32,
// double-buffered async staging.
// ---------------------------------------------------------------------------
__global__ __launch_bounds__(256) void energy_kernel(
    const unsigned short* __restrict__ Kt,
    const unsigned short* __restrict__ Qt,
    float* __restrict__ Et)
{
    const int b   = blockIdx.z;
    const int m0  = blockIdx.y * 128;   // s tile
    const int n0  = blockIdx.x * 64;    // t tile
    const int tid = threadIdx.x;
    const int lane = tid & 31, wave = tid >> 5;
    const int wm = wave & 3, wn = wave >> 2;
    const int half = lane >> 4, nl = lane & 15;

    __shared__ unsigned short As[2][128 * 32];
    __shared__ unsigned short Bs[2][64 * 32];

    const int rA = tid >> 1, cA = (tid & 1) * 16;
    const int rB = tid >> 2, cB = (tid & 3) * 8;
    const unsigned short* srcA =
        Kt + (size_t)b * SS * INC + (size_t)(m0 + rA) * INC + cA;
    const unsigned short* srcB =
        Qt + (size_t)b * SS * INC + (size_t)(n0 + rB) * INC + cB;

    f32x8 acc[2][2] = {};

    cp16_g2l(&As[0][rA * 32 + cA],     srcA);
    cp16_g2l(&As[0][rA * 32 + cA + 8], srcA + 8);
    cp16_g2l(&Bs[0][rB * 32 + cB],     srcB);

    for (int k0 = 0; k0 < INC; k0 += 32) {
        const int buf = (k0 >> 5) & 1;
        if (k0 + 32 < INC) {
            const int nb = buf ^ 1;
            cp16_g2l(&As[nb][rA * 32 + cA],     srcA + k0 + 32);
            cp16_g2l(&As[nb][rA * 32 + cA + 8], srcA + k0 + 40);
            cp16_g2l(&Bs[nb][rB * 32 + cB],     srcB + k0 + 32);
            async_wait<3>();
        } else {
            async_wait<0>();
        }
        __syncthreads();

        bf16x16 afrag[2];
#pragma unroll
        for (int mi = 0; mi < 2; ++mi)
            afrag[mi] = load_frag_a(&As[buf][(wm * 32 + mi * 16 + nl) * 32], half);
#pragma unroll
        for (int ni = 0; ni < 2; ++ni) {
            bf16x16 bfrag =
                load_frag_b(&Bs[buf][(wn * 32 + ni * 16 + nl) * 32], half);
#pragma unroll
            for (int mi = 0; mi < 2; ++mi)
                acc[mi][ni] = __builtin_amdgcn_wmma_f32_16x16x32_bf16(
                    false, afrag[mi], false, bfrag, (short)0,
                    acc[mi][ni], false, false);
        }
        __syncthreads();
    }

    // Store Et[t][s]: lane has fixed t, 8 consecutive s -> contiguous 32B f32
#pragma unroll
    for (int ni = 0; ni < 2; ++ni) {
        const int t = n0 + wn * 32 + ni * 16 + nl;
#pragma unroll
        for (int mi = 0; mi < 2; ++mi) {
            const int s = m0 + wm * 32 + mi * 16 + half * 8;
            *(f32x8*)(Et + ((size_t)b * SS + t) * SS + s) = acc[mi][ni];
        }
    }
}

// ---------------------------------------------------------------------------
// Kernel 3: per (b,s): online max + sum_t exp(E[s,t]-max). Column walk of Et.
// ---------------------------------------------------------------------------
__global__ __launch_bounds__(512) void rowstats_kernel(
    const float* __restrict__ Et, float* __restrict__ rmax, float* __restrict__ rden)
{
    const int b = blockIdx.x;
    const int s = threadIdx.x;
    const float* base = Et + (size_t)b * SS * SS + s;
    float m = -3.0e38f, d = 0.0f;
    for (int t = 0; t < SS; ++t) {
        const float v = base[(size_t)t * SS];
        const float nm = fmaxf(m, v);
        d = d * __expf(m - nm) + __expf(v - nm);
        m = nm;
    }
    rmax[b * SS + s] = m;
    rden[b * SS + s] = d;
}

// ---------------------------------------------------------------------------
// Kernel 4: u[b][t] = sum_s (Wout[s]/den[s]) * exp(Et[t][s] - max[s]).
// One wave32 per (b,t); lane-strided over s; shuffle reduction.
// ---------------------------------------------------------------------------
__global__ __launch_bounds__(256) void u_kernel(
    const float* __restrict__ Et, const float* __restrict__ rmax,
    const float* __restrict__ rden, const float* __restrict__ Wout,
    float* __restrict__ u)
{
    const int gw   = (int)((blockIdx.x * blockDim.x + threadIdx.x) >> 5);
    const int lane = threadIdx.x & 31;
    const int b = gw >> 9;          // 512 rows per batch
    const int t = gw & 511;
    const float* row = Et + ((size_t)b * SS + t) * SS;
    const float* mx  = rmax + b * SS;
    const float* dn  = rden + b * SS;
    float acc = 0.0f;
    for (int s = lane; s < SS; s += 32)
        acc += (Wout[s] / dn[s]) * __expf(row[s] - mx[s]);
#pragma unroll
    for (int off = 16; off > 0; off >>= 1) acc += __shfl_xor(acc, off, 32);
    if (lane == 0) u[b * SS + t] = acc;
}

// ---------------------------------------------------------------------------
// Kernel 5: out[b][i] = bout + sum_t Vt[b][t][i] * u[b][t]   (coalesced over i)
// ---------------------------------------------------------------------------
__global__ __launch_bounds__(256) void out_kernel(
    const unsigned short* __restrict__ Vt, const float* __restrict__ u,
    const float* __restrict__ bout, float* __restrict__ out)
{
    const int b = blockIdx.y;
    const int i = blockIdx.x * 256 + threadIdx.x;
    const unsigned short* v = Vt + (size_t)b * SS * INC + i;
    const float* ub = u + b * SS;
    float acc = bout[0];
    for (int t = 0; t < SS; ++t)
        acc += ub[t] * bf2f(v[(size_t)t * INC]);
    out[b * INC + i] = acc;
}

// ---------------------------------------------------------------------------
extern "C" void kernel_launch(void* const* d_in, const int* in_sizes, int n_in,
                              void* d_out, int out_size, void* d_ws, size_t ws_size,
                              hipStream_t stream)
{
    const float* x    = (const float*)d_in[0];
    const float* Wq   = (const float*)d_in[1];
    const float* bq   = (const float*)d_in[2];
    const float* Wk   = (const float*)d_in[3];
    const float* bk   = (const float*)d_in[4];
    const float* Wv   = (const float*)d_in[5];
    const float* bv   = (const float*)d_in[6];
    const float* Wout = (const float*)d_in[7];
    const float* bout = (const float*)d_in[8];
    float* out = (float*)d_out;

    char* ws = (char*)d_ws;
    const size_t XB  = (size_t)BB * INC * SS * sizeof(unsigned short);  // 64 MB
    const size_t WB  = (size_t)SS * SS * sizeof(unsigned short);        // 512 KB
    const size_t EB  = (size_t)BB * SS * SS * sizeof(float);            // 32 MB
    unsigned short* xb  = (unsigned short*)(ws);
    unsigned short* Qt  = (unsigned short*)(ws + XB);
    unsigned short* Kt  = (unsigned short*)(ws + 2 * XB);
    unsigned short* Vt  = (unsigned short*)(ws + 3 * XB);
    float*          Et  = (float*)(ws + 4 * XB);
    unsigned short* Wqb = (unsigned short*)(ws + 4 * XB + EB);
    unsigned short* Wkb = (unsigned short*)(ws + 4 * XB + EB + WB);
    unsigned short* Wvb = (unsigned short*)(ws + 4 * XB + EB + 2 * WB);
    float* rmax = (float*)(ws + 4 * XB + EB + 3 * WB);
    float* rden = rmax + (size_t)BB * SS;
    float* u    = rden + (size_t)BB * SS;

    // 0) bulk f32 -> bf16 conversion
    cvt_kernel<<<(BB * INC * SS) / 1024, 256, 0, stream>>>(x, xb);
    cvt_kernel<<<(SS * SS) / 1024, 256, 0, stream>>>(Wq, Wqb);
    cvt_kernel<<<(SS * SS) / 1024, 256, 0, stream>>>(Wk, Wkb);
    cvt_kernel<<<(SS * SS) / 1024, 256, 0, stream>>>(Wv, Wvb);

    // 1) fused QKV projection (transposed bf16 outputs)
    dim3 g1(SS / 64, INC / 128, BB);     // (8, 16, 32)
    qkv_kernel<<<g1, 256, 0, stream>>>(xb, Wqb, Wkb, Wvb, bq, bk, bv, Qt, Kt, Vt);

    // 2) energy (K^T Q, contraction over INC)
    dim3 g2(SS / 64, SS / 128, BB);      // (8, 4, 32)
    energy_kernel<<<g2, 256, 0, stream>>>(Kt, Qt, Et);

    // 3) softmax row stats
    rowstats_kernel<<<BB, 512, 0, stream>>>(Et, rmax, rden);

    // 4) u = A^T w  (folded rank-1 output projection)
    u_kernel<<<(BB * SS) / 8, 256, 0, stream>>>(Et, rmax, rden, Wout, u);

    // 5) out = V u + bout
    dim3 g5(INC / 256, BB);
    out_kernel<<<g5, 256, 0, stream>>>(Vt, u, bout, out);
}
